// STDecoder_35433480192160
// MI455X (gfx1250) — matrix operand order repaired and checked by reading, not compiled, not run on hardware
//
#include <hip/hip_runtime.h>

typedef __bf16 bf16;
typedef __attribute__((ext_vector_type(16))) __bf16       v16bf;
typedef __attribute__((ext_vector_type(8)))  float        v8f;
typedef __attribute__((ext_vector_type(4)))  unsigned int v4u;
typedef __attribute__((ext_vector_type(8)))  unsigned int v8u;
typedef __attribute__((ext_vector_type(4)))  unsigned int u32x4;
typedef __attribute__((ext_vector_type(4)))  int          i32x4;
typedef __attribute__((ext_vector_type(8)))  int          i32x8;
typedef __attribute__((address_space(3)))    __bf16       lds_bf16;

#define VOCAB  32000
#define EMBED  512
#define DECODE 1024
#define BATCH  64
#define TSTEPS 19                 // output timesteps
#define NCHUNK 128                // logits N tile
#define NCH    (VOCAB / NCHUNK)   // 250 chunks per row

#if __has_builtin(__builtin_amdgcn_tensor_load_to_lds)
#define USE_TDM 1
#endif

// ---------- helpers ----------
static __device__ __forceinline__ bf16 f2bf(float f) {
  unsigned int u = __builtin_bit_cast(unsigned int, f);
  u += 0x7FFFu + ((u >> 16) & 1u);              // round-to-nearest-even
  unsigned short s = (unsigned short)(u >> 16);
  return __builtin_bit_cast(bf16, s);
}
static __device__ __forceinline__ float sigm(float x) { return 1.0f / (1.0f + __expf(-x)); }

// A fragment: 16x32 bf16, row-major source [m][k], ld in elements.
// lane<16 holds row m=lane, K = k0..k0+15; lane>=16: K = k0+16..k0+31.
static __device__ __forceinline__ v16bf load_a(const bf16* A, int ld, int k0, int lane) {
  int m = lane & 15, hi = lane >> 4;
  const bf16* p = A + (size_t)m * ld + k0 + 16 * hi;
  v4u x0 = *(const v4u*)p;
  v4u x1 = *(const v4u*)(p + 8);
  v8u r = {x0.x, x0.y, x0.z, x0.w, x1.x, x1.y, x1.z, x1.w};
  return __builtin_bit_cast(v16bf, r);
}
// B fragment: 32x16 bf16 where B[k][n] = W[n][k], W row-major [n][k].
// lane = n + 16*(k/16); elements = 16 consecutive K of one weight row.
static __device__ __forceinline__ v16bf load_b(const bf16* Wrow, int ld, int k0, int lane) {
  int n = lane & 15, kg = lane >> 4;
  const bf16* p = Wrow + (size_t)n * ld + k0 + 16 * kg;
  v4u x0 = *(const v4u*)p;
  v4u x1 = *(const v4u*)(p + 8);
  v8u r = {x0.x, x0.y, x0.z, x0.w, x1.x, x1.y, x1.z, x1.w};
  return __builtin_bit_cast(v16bf, r);
}
// B fragment from an LDS tile laid out [rows][32] bf16 (row = n-local).
static __device__ __forceinline__ v16bf load_b_lds(const bf16* tile32, int nbase, int lane) {
  int n = lane & 15, kg = lane >> 4;
  const bf16* p = tile32 + (size_t)(nbase + n) * 32 + 16 * kg;
  v4u x0 = *(const v4u*)p;
  v4u x1 = *(const v4u*)(p + 8);
  v8u r = {x0.x, x0.y, x0.z, x0.w, x1.x, x1.y, x1.z, x1.w};
  return __builtin_bit_cast(v16bf, r);
}
#define WMMA_BF16(a, b, c) \
  __builtin_amdgcn_wmma_f32_16x16x32_bf16(false, (a), false, (b), (short)0, (c), false, false)

// ---------- fp32 -> bf16 conversion ----------
__global__ void k_f2bf(const float* __restrict__ src, bf16* __restrict__ dst, int n) {
  int i = blockIdx.x * blockDim.x + threadIdx.x;
  if (i < n) dst[i] = f2bf(src[i]);
}

// ---------- embedding gather (padding_idx = 0) into X slots 1..19 ----------
__global__ void k_embed(const int* __restrict__ cap, const float* __restrict__ E,
                        bf16* __restrict__ X) {
  int i = blockIdx.x * blockDim.x + threadIdx.x;
  if (i >= TSTEPS * BATCH * EMBED) return;
  int t = i / (BATCH * EMBED);
  int r = i % (BATCH * EMBED);
  int b = r / EMBED, k = r % EMBED;
  int tok = cap[b * 20 + t];
  float v = (tok == 0) ? 0.0f : E[(size_t)tok * EMBED + k];
  X[(size_t)(1 + t) * BATCH * EMBED + (size_t)b * EMBED + k] = f2bf(v);
}

__global__ void k_zero(float* __restrict__ p, int n) {
  int i = blockIdx.x * blockDim.x + threadIdx.x;
  if (i < n) p[i] = 0.0f;
}

// ---------- x0 = joint @ Wproj^T + bproj  (64x512, K=1024), bf16 out ----------
__global__ void __launch_bounds__(256) k_proj(const bf16* __restrict__ J,
                                              const bf16* __restrict__ Wp,
                                              const float* __restrict__ bp,
                                              bf16* __restrict__ X0) {
  int tid = threadIdx.x, lane = tid & 31, w = tid >> 5;
  int n0 = blockIdx.x * 64;
  int mt  = w >> 1;                 // 0..3
  int nt0 = (w * 2) & 3, nt1 = (w * 2 + 1) & 3;
  v8f acc0 = {}, acc1 = {};
  const bf16* Arow = J + (size_t)mt * 16 * DECODE;
  const bf16* B0 = Wp + (size_t)(n0 + nt0 * 16) * DECODE;
  const bf16* B1 = Wp + (size_t)(n0 + nt1 * 16) * DECODE;
  for (int k0 = 0; k0 < DECODE; k0 += 32) {
    v16bf a  = load_a(Arow, DECODE, k0, lane);
    v16bf b0 = load_b(B0,   DECODE, k0, lane);
    v16bf b1 = load_b(B1,   DECODE, k0, lane);
    acc0 = WMMA_BF16(a, b0, acc0);
    acc1 = WMMA_BF16(a, b1, acc1);
  }
  int nl = lane & 15, hi = lane >> 4;
  for (int r = 0; r < 8; ++r) {
    int m = mt * 16 + r + 8 * hi;
    int na = n0 + nt0 * 16 + nl;
    int nb = n0 + nt1 * 16 + nl;
    X0[(size_t)m * EMBED + na] = f2bf(acc0[r] + bp[na]);
    X0[(size_t)m * EMBED + nb] = f2bf(acc1[r] + bp[nb]);
  }
}

// ---------- fused LSTM step: gates GEMM + pointwise cell ----------
// 32 blocks x 256 threads; block handles 32 d-columns across all 4 gates.
__global__ void __launch_bounds__(256) k_lstm(const bf16* __restrict__ Xt,
                                              const bf16* __restrict__ Hprev,
                                              const bf16* __restrict__ Wih,
                                              const bf16* __restrict__ Whh,
                                              const float* __restrict__ bih,
                                              const float* __restrict__ bhh,
                                              float* __restrict__ C,
                                              bf16* __restrict__ Hout, int use_h) {
  __shared__ float gbuf[4][64][32];     // [gate][m][d_local]  32 KB
  int tid = threadIdx.x, lane = tid & 31, w = tid >> 5;
  int d0 = blockIdx.x * 32;
  v8f acc[4] = {};
  int gate[4], mt[4], nt[4];
#pragma unroll
  for (int j = 0; j < 4; ++j) {
    int s = w * 4 + j;                  // 0..31 subtiles: 4 gates x 4 mt x 2 nt
    gate[j] = s >> 3; int rem = s & 7; mt[j] = rem >> 1; nt[j] = rem & 1;
  }
  for (int k0 = 0; k0 < EMBED; k0 += 32) {
#pragma unroll
    for (int j = 0; j < 4; ++j) {
      v16bf a = load_a(Xt + (size_t)mt[j] * 16 * EMBED, EMBED, k0, lane);
      int ng = gate[j] * DECODE + d0 + nt[j] * 16;
      v16bf b = load_b(Wih + (size_t)ng * EMBED, EMBED, k0, lane);
      acc[j] = WMMA_BF16(a, b, acc[j]);
    }
  }
  if (use_h) {
    for (int k0 = 0; k0 < DECODE; k0 += 32) {
#pragma unroll
      for (int j = 0; j < 4; ++j) {
        v16bf a = load_a(Hprev + (size_t)mt[j] * 16 * DECODE, DECODE, k0, lane);
        int ng = gate[j] * DECODE + d0 + nt[j] * 16;
        v16bf b = load_b(Whh + (size_t)ng * DECODE, DECODE, k0, lane);
        acc[j] = WMMA_BF16(a, b, acc[j]);
      }
    }
  }
  int nl = lane & 15, hi = lane >> 4;
#pragma unroll
  for (int j = 0; j < 4; ++j)
#pragma unroll
    for (int r = 0; r < 8; ++r)
      gbuf[gate[j]][mt[j] * 16 + r + 8 * hi][nt[j] * 16 + nl] = acc[j][r];
  __syncthreads();
  for (int p = tid; p < 64 * 32; p += 256) {
    int m = p >> 5, dl = p & 31, d = d0 + dl;
    float iv = gbuf[0][m][dl] + bih[d]              + bhh[d];
    float fv = gbuf[1][m][dl] + bih[DECODE + d]     + bhh[DECODE + d];
    float gv = gbuf[2][m][dl] + bih[2 * DECODE + d] + bhh[2 * DECODE + d];
    float ov = gbuf[3][m][dl] + bih[3 * DECODE + d] + bhh[3 * DECODE + d];
    float c  = C[(size_t)m * DECODE + d];           // pre-zeroed for step 0
    float cn = sigm(fv) * c + sigm(iv) * tanhf(gv);
    C[(size_t)m * DECODE + d] = cn;
    Hout[(size_t)m * DECODE + d] = f2bf(sigm(ov) * tanhf(cn));
  }
}

// ---------- batched logits GEMM (1216x32000, K=1024) + softmax partials ----------
// grid = (250 n-chunks, 19 timesteps); block computes 64x128 tile.
// W_out tiles (128 rows x 32 K, 8 KB bf16) are DMA'd into LDS by the Tensor
// Data Mover, double-buffered, overlapped with WMMA on the previous tile.
__global__ void __launch_bounds__(256) k_logits(const bf16* __restrict__ H1,
                                                const bf16* __restrict__ Wout,
                                                const float* __restrict__ bout,
                                                float* __restrict__ out,
                                                float* __restrict__ pmax,
                                                float* __restrict__ psum) {
  __shared__ float tile[64][128];                  // 32 KB
  __shared__ bf16  btile[2][NCHUNK][32];           // 16 KB, double-buffered B tile
  int tid = threadIdx.x, lane = tid & 31, w = tid >> 5;
  int nc = blockIdx.x, mb = blockIdx.y;
  int n0 = nc * NCHUNK;
  const bf16* A = H1 + (size_t)mb * BATCH * DECODE;
  int mt = w & 3;
  int ntb = (w >> 2) * 4;
  v8f acc[4] = {};
  const bf16* Arow = A + (size_t)mt * 16 * DECODE;
  constexpr int KSTEPS = DECODE / 32;

  auto issue_stage = [&](int k0, int buf) {
#ifdef USE_TDM
    if (w == 0) {
      // TDM descriptor (ISA 8.3/8.4): 2D tensor, data_size=2B,
      // tile = 32 (K, contiguous) x 128 (rows), row stride = DECODE.
      unsigned lds_addr = (unsigned)
          reinterpret_cast<unsigned long long>((lds_bf16*)&btile[buf][0][0]);
      unsigned long long ga =
          reinterpret_cast<unsigned long long>(Wout + (size_t)n0 * DECODE + k0);
      u32x4 g0 = { 1u,                                   // count = 1 valid D#
                   lds_addr,                             // lds_addr
                   (unsigned)ga,                         // global_addr[31:0]
                   (unsigned)(ga >> 32) | 0x80000000u }; // addr[56:32] | type=2
      i32x8 g1 = { (int)0x00010000u,                     // data_size = 2 bytes
                   (int)(1024u << 16),                   // tensor_dim0 = 1024 (lo16 in w1 hi)
                   (int)0x7D000000u,                     // tensor_dim1 = 32000 (lo16 in w2 hi)
                   (int)(32u << 16),                     // tile_dim0 = 32
                   128,                                  // tile_dim1 = 128
                   1024,                                 // tensor_dim0_stride = 1024
                   0, 0 };
      i32x4 z4 = {0, 0, 0, 0};
#if __clang_major__ >= 23
      i32x8 z8 = {0, 0, 0, 0, 0, 0, 0, 0};
      __builtin_amdgcn_tensor_load_to_lds(g0, g1, z4, z4, z8, 0);
#else
      __builtin_amdgcn_tensor_load_to_lds(g0, g1, z4, z4, 0);
#endif
    }
#else
    // Fallback: cooperative staging, 512 x 16B chunks, 2 per thread.
    for (int c = tid; c < 512; c += 256) {
      int rrow = c >> 2, kc = c & 3;
      const v4u* gp = (const v4u*)(Wout + (size_t)(n0 + rrow) * DECODE + k0 + kc * 8);
      *(v4u*)&btile[buf][rrow][kc * 8] = *gp;
    }
#endif
  };
  auto wait_stage = [&]() {
#ifdef USE_TDM
#if __has_builtin(__builtin_amdgcn_s_wait_tensorcnt)
    if (w == 0) __builtin_amdgcn_s_wait_tensorcnt(0);
#endif
#endif
  };

  issue_stage(0, 0);
  wait_stage();
  __syncthreads();

  for (int ks = 0; ks < KSTEPS; ++ks) {
    int cur = ks & 1;
    if (ks + 1 < KSTEPS) issue_stage((ks + 1) * 32, cur ^ 1);
    int k0 = ks * 32;
    v16bf a = load_a(Arow, DECODE, k0, lane);
#pragma unroll
    for (int j = 0; j < 4; ++j) {
      v16bf b = load_b_lds(&btile[cur][0][0], (ntb + j) * 16, lane);
      acc[j] = WMMA_BF16(a, b, acc[j]);
    }
    wait_stage();
    __syncthreads();
  }

  int nl = lane & 15, hi = lane >> 4;
#pragma unroll
  for (int j = 0; j < 4; ++j) {
    int col = (ntb + j) * 16 + nl;
    float bo = bout[n0 + col];
#pragma unroll
    for (int r = 0; r < 8; ++r)
      tile[mt * 16 + r + 8 * hi][col] = acc[j][r] + bo;
  }
  __syncthreads();
  // raw logits -> d_out  (out layout [b][t][v], b == m)
  for (int p = tid; p < 64 * 128; p += 256) {
    int m = p >> 7, col = p & 127;
    out[((size_t)m * TSTEPS + mb) * VOCAB + n0 + col] = tile[m][col];
  }
  // per-row chunk max / sum-exp partials
  if (tid < 64) {
    int m = tid;
    float mx = -3.0e38f;
    for (int c = 0; c < NCHUNK; ++c) mx = fmaxf(mx, tile[m][c]);
    float s = 0.0f;
    for (int c = 0; c < NCHUNK; ++c) s += __expf(tile[m][c] - mx);
    int rg = mb * BATCH + m;
    pmax[(size_t)rg * NCH + nc] = mx;
    psum[(size_t)rg * NCH + nc] = s;
  }
}

__global__ void k_rowred(const float* __restrict__ pmax, const float* __restrict__ psum,
                         float* __restrict__ rls) {
  int r = blockIdx.x * blockDim.x + threadIdx.x;
  if (r >= TSTEPS * BATCH) return;
  float M = -3.0e38f;
  for (int c = 0; c < NCH; ++c) M = fmaxf(M, pmax[(size_t)r * NCH + c]);
  float S = 0.0f;
  for (int c = 0; c < NCH; ++c) S += psum[(size_t)r * NCH + c] * __expf(pmax[(size_t)r * NCH + c] - M);
  rls[r] = M + __logf(S);
}

__global__ void k_sub(float* __restrict__ out, const float* __restrict__ rls) {
  size_t i = (size_t)blockIdx.x * blockDim.x + threadIdx.x;
  if (i >= (size_t)BATCH * TSTEPS * VOCAB) return;
  int b = (int)(i / (TSTEPS * VOCAB));
  int t = (int)((i % (TSTEPS * VOCAB)) / VOCAB);
  out[i] -= rls[t * BATCH + b];
}

// ---------- launcher ----------
extern "C" void kernel_launch(void* const* d_in, const int* in_sizes, int n_in,
                              void* d_out, int out_size, void* d_ws, size_t ws_size,
                              hipStream_t stream) {
  (void)in_sizes; (void)n_in; (void)out_size; (void)ws_size;
  const float* joint = (const float*)d_in[0];
  const int*   cap   = (const int*)d_in[1];
  const float* E     = (const float*)d_in[4];
  const float* Wproj = (const float*)d_in[5];
  const float* bproj = (const float*)d_in[6];
  const float* Wih   = (const float*)d_in[7];
  const float* Whh   = (const float*)d_in[8];
  const float* bih   = (const float*)d_in[9];
  const float* bhh   = (const float*)d_in[10];
  const float* Wout  = (const float*)d_in[11];
  const float* bout  = (const float*)d_in[12];
  float* out = (float*)d_out;

  char* ws = (char*)d_ws;
  size_t off = 0;
  auto carve = [&](size_t bytes) {
    char* p = ws + off;
    off += (bytes + 255) & ~(size_t)255;
    return p;
  };
  bf16*  wih_b  = (bf16*)carve((size_t)4 * DECODE * EMBED * 2);
  bf16*  whh_b  = (bf16*)carve((size_t)4 * DECODE * DECODE * 2);
  bf16*  wout_b = (bf16*)carve((size_t)VOCAB * DECODE * 2);
  bf16*  wpr_b  = (bf16*)carve((size_t)EMBED * DECODE * 2);
  bf16*  jnt_b  = (bf16*)carve((size_t)BATCH * DECODE * 2);
  bf16*  Xbuf   = (bf16*)carve((size_t)20 * BATCH * EMBED * 2);   // slot 0 = x0, 1..19 = emb
  bf16*  Hbuf   = (bf16*)carve((size_t)20 * BATCH * DECODE * 2);  // h after step s
  float* Cbuf   = (float*)carve((size_t)BATCH * DECODE * 4);
  float* pmax   = (float*)carve((size_t)TSTEPS * BATCH * NCH * 4);
  float* psum   = (float*)carve((size_t)TSTEPS * BATCH * NCH * 4);
  float* rls    = (float*)carve((size_t)TSTEPS * BATCH * 4);

  auto cvt = [&](const float* s, bf16* d, int n) {
    k_f2bf<<<(n + 255) / 256, 256, 0, stream>>>(s, d, n);
  };
  cvt(Wih,   wih_b,  4 * DECODE * EMBED);
  cvt(Whh,   whh_b,  4 * DECODE * DECODE);
  cvt(Wout,  wout_b, VOCAB * DECODE);
  cvt(Wproj, wpr_b,  EMBED * DECODE);
  cvt(joint, jnt_b,  BATCH * DECODE);

  k_embed<<<(TSTEPS * BATCH * EMBED + 255) / 256, 256, 0, stream>>>(cap, E, Xbuf);
  k_zero<<<(BATCH * DECODE + 255) / 256, 256, 0, stream>>>(Cbuf, BATCH * DECODE);
  k_proj<<<EMBED / 64, 256, 0, stream>>>(jnt_b, wpr_b, bproj, Xbuf);  // writes slot 0

  for (int s = 0; s < 20; ++s) {
    const bf16* Xt = Xbuf + (size_t)s * BATCH * EMBED;
    const bf16* Hp = Hbuf + (size_t)(s > 0 ? s - 1 : 0) * BATCH * DECODE;
    bf16* Ho = Hbuf + (size_t)s * BATCH * DECODE;
    k_lstm<<<DECODE / 32, 256, 0, stream>>>(Xt, Hp, wih_b, whh_b, bih, bhh, Cbuf, Ho,
                                            s > 0 ? 1 : 0);
  }

  k_logits<<<dim3(NCH, TSTEPS), 256, 0, stream>>>(Hbuf + (size_t)BATCH * DECODE,
                                                  wout_b, bout, out, pmax, psum);
  k_rowred<<<(TSTEPS * BATCH + 255) / 256, 256, 0, stream>>>(pmax, psum, rls);
  size_t tot = (size_t)BATCH * TSTEPS * VOCAB;
  k_sub<<<(unsigned)((tot + 255) / 256), 256, 0, stream>>>(out, rls);
}
